// CRNN_Michael_30889404793207
// MI455X (gfx1250) — matrix operand order repaired and checked by reading, not compiled, not run on hardware
//
#include <hip/hip_runtime.h>
#include <hip/hip_bf16.h>

// ---------------------------------------------------------------------------
// CDNA5 (gfx1250) CRNN: conv encoder -> 3x BiLSTM (WMMA) -> attn decoder (WMMA)
// wave32, v_wmma_f32_16x16x32_bf16 for every matmul. All WMMA weight operands
// pre-converted to bf16 once (L2-resident) so serial loops have no converts.
// ---------------------------------------------------------------------------

typedef __attribute__((ext_vector_type(16))) __bf16 v16bf;
typedef __attribute__((ext_vector_type(8)))  float  v8f;

#define DEV __device__ __forceinline__

DEV float sigf(float x) { return 1.f / (1.f + __expf(-x)); }

// A-fragment (16x32 bf16, row-major source, lda in elements).
// Contiguous k order (matches load_bfrag): lanes 0-15 row M=lane, K = 0..15; lanes 16-31 K = 16..31.
DEV v16bf load_afrag(const __bf16* A, int lda, int m0, int k0, int lane) {
  const int r  = m0 + (lane & 15);
  const int kb = k0 + ((lane & 16) ? 16 : 0);
  const __bf16* p = A + (size_t)r * lda + kb;
  v16bf a;
#pragma unroll
  for (int e = 0; e < 8; ++e)  a[e] = p[e];
#pragma unroll
  for (int e = 8; e < 16; ++e) a[e] = p[e];
  return a;
}

// B-fragment (32x16) from pre-converted bf16 weight Wb[N][K] (row-major):
// B[k][n] = Wb[n][k]. Lanes 0-15: col n, K=0..15; lanes 16-31: col n, K=16..31.
// Per-lane data is 16 contiguous bf16 (32B, aligned) -> two global_load_b128.
DEV v16bf load_bfrag(const __bf16* Wb, int ldw, int n0, int k0, int lane) {
  const int n  = n0 + (lane & 15);
  const int kb = k0 + ((lane & 16) ? 16 : 0);
  return *(const v16bf*)(Wb + (size_t)n * ldw + kb);
}

#define WMMA_BF16(a, b, c) \
  __builtin_amdgcn_wmma_f32_16x16x32_bf16(false, (a), false, (b), (short)0, (c), false, false)

// ---------------------------------------------------------------------------
// Weight conversion: f32 -> bf16 (optionally zero-padded rows for out_W)
// ---------------------------------------------------------------------------
__global__ void k_cvt_bf16(const float* __restrict__ src, __bf16* __restrict__ dst, int n) {
  int idx = blockIdx.x * blockDim.x + threadIdx.x;
  if (idx < n) dst[idx] = (__bf16)src[idx];
}

__global__ void k_cvt_bf16_pad(const float* __restrict__ src, __bf16* __restrict__ dst,
                               int rows_real, int cols, int rows_pad) {
  int idx = blockIdx.x * blockDim.x + threadIdx.x;
  if (idx >= rows_pad * cols) return;
  int row = idx / cols;
  dst[idx] = (row < rows_real) ? (__bf16)src[idx] : (__bf16)0.f;
}

// ---------------------------------------------------------------------------
// Generic GEMM: C[M,N](f32) = A[M,K](bf16) @ Wb[N,K]^T + bias. One wave/tile.
// grid.x = (M/16)*(N/16), block = 32.
// ---------------------------------------------------------------------------
__global__ __launch_bounds__(32) void k_gemm_bf16(
    const __bf16* __restrict__ A, const __bf16* __restrict__ Wb,
    const float* __restrict__ bias, float* __restrict__ C, int N, int K) {
  const int lane = threadIdx.x & 31;
  const int tiles_n = N >> 4;
  const int tm = blockIdx.x / tiles_n, tn = blockIdx.x % tiles_n;
  const int m0 = tm * 16, n0 = tn * 16;
  v8f acc = {};
  for (int k0 = 0; k0 < K; k0 += 32) {
    __builtin_prefetch(A + (size_t)(m0 + (lane & 15)) * K + k0 + 32, 0, 0);
    v16bf a = load_afrag(A, K, m0, k0, lane);
    v16bf b = load_bfrag(Wb, K, n0, k0, lane);
    acc = WMMA_BF16(a, b, acc);
  }
  const int n  = n0 + (lane & 15);
  const int mo = m0 + ((lane & 16) ? 8 : 0);
  const float bv = bias ? bias[n] : 0.f;
#pragma unroll
  for (int v = 0; v < 8; ++v) C[(size_t)(mo + v) * N + n] = acc[v] + bv;
}

// ---------------------------------------------------------------------------
// CNN encoder kernels (VALU; tiny fraction of total work)
// ---------------------------------------------------------------------------
__global__ void k_conv1(const float* __restrict__ x, const float* __restrict__ w,
                        const float* __restrict__ bias, float* __restrict__ out) {
  int idx = blockIdx.x * blockDim.x + threadIdx.x;
  const int total = 64 * 8 * 15 * 511;
  if (idx >= total) return;
  int ow = idx % 511; int t = idx / 511;
  int oh = t % 15; t /= 15;
  int oc = t % 8;  int b = t / 8;
  const int ih0 = oh * 4, iw0 = ow * 2;
  float s = bias[oc];
  for (int ic = 0; ic < 3; ++ic)
    for (int kh = 0; kh < 6; ++kh)
      for (int kw = 0; kw < 4; ++kw)
        s += x[(((size_t)b * 3 + ic) * 64 + ih0 + kh) * 1024 + iw0 + kw] *
             w[((oc * 3 + ic) * 6 + kh) * 4 + kw];
  out[idx] = s > 0.f ? s : 0.01f * s;
}

__global__ void k_conv2(const float* __restrict__ c1, const float* __restrict__ w,
                        const float* __restrict__ bias, float* __restrict__ out) {
  int idx = blockIdx.x * blockDim.x + threadIdx.x;
  const int total = 64 * 32 * 10 * 508;
  if (idx >= total) return;
  int ow = idx % 508; int t = idx / 508;
  int oh = t % 10; t /= 10;
  int oc = t % 32; int b = t / 32;
  float s = bias[oc];
  for (int ic = 0; ic < 8; ++ic)
    for (int kh = 0; kh < 6; ++kh)
      for (int kw = 0; kw < 4; ++kw)
        s += c1[(((size_t)b * 8 + ic) * 15 + oh + kh) * 511 + ow + kw] *
             w[((oc * 8 + ic) * 6 + kh) * 4 + kw];
  out[idx] = s > 0.f ? s : 0.01f * s;
}

__global__ void k_pool1(const float* __restrict__ c2, float* __restrict__ out) {
  int idx = blockIdx.x * blockDim.x + threadIdx.x;
  const int total = 64 * 32 * 2 * 254;
  if (idx >= total) return;
  int ow = idx % 254; int t = idx / 254;
  int oh = t % 2; t /= 2;
  int oc = t % 32; int b = t / 32;
  float m = -3.0e38f;
  for (int kh = 0; kh < 4; ++kh)
    for (int kw = 0; kw < 2; ++kw)
      m = fmaxf(m, c2[(((size_t)b * 32 + oc) * 10 + oh * 4 + kh) * 508 + ow * 2 + kw]);
  out[idx] = m;
}

__global__ void k_conv3(const float* __restrict__ p1, const float* __restrict__ w,
                        const float* __restrict__ bias, float* __restrict__ out) {
  int idx = blockIdx.x * blockDim.x + threadIdx.x;
  const int total = 64 * 64 * 2 * 254;
  if (idx >= total) return;
  int ow = idx % 254; int t = idx / 254;
  int oh = t % 2; t /= 2;
  int oc = t % 64; int b = t / 64;
  float s = bias[oc];
  for (int ic = 0; ic < 32; ++ic)
    for (int kh = -1; kh <= 1; ++kh) {
      int ih = oh + kh;
      if (ih < 0 || ih > 1) continue;
      for (int kw = -1; kw <= 1; ++kw) {
        int iw = ow + kw;
        if (iw < 0 || iw > 253) continue;
        s += p1[(((size_t)b * 32 + ic) * 2 + ih) * 254 + iw] *
             w[((oc * 32 + ic) * 3 + (kh + 1)) * 3 + (kw + 1)];
      }
    }
  out[idx] = s > 0.f ? s : 0.01f * s;
}

// pool (1,2)/(1,2) + reshape/transpose -> X0[s][b][f] bf16, f = ch*2 + h
__global__ void k_pool2_repack(const float* __restrict__ c3, __bf16* __restrict__ X0) {
  int idx = blockIdx.x * blockDim.x + threadIdx.x;
  const int total = 64 * 64 * 2 * 127;
  if (idx >= total) return;
  int s = idx % 127; int t = idx / 127;
  int h = t & 1; t >>= 1;
  int ch = t % 64; int b = t / 64;
  const float* p = c3 + (((size_t)b * 64 + ch) * 2 + h) * 254 + 2 * s;
  X0[((size_t)s * 64 + b) * 128 + ch * 2 + h] = (__bf16)fmaxf(p[0], p[1]);
}

// ---------------------------------------------------------------------------
// Persistent recurrent LSTM layer (one direction per block; grid=2).
// G[64,512] = Xproj[s] + h @ Whh^T via WMMA; state in LDS (48 KB).
// ---------------------------------------------------------------------------
__global__ __launch_bounds__(512) void k_lstm_layer(
    const float* __restrict__ XpF, const float* __restrict__ XpB,
    const __bf16* __restrict__ WhhF, const __bf16* __restrict__ WhhB,
    __bf16* __restrict__ Xnext) {
  __shared__ float  cst[64 * 128];
  __shared__ __bf16 hsh[64 * 128];
  const int S = 127;
  const int dir = blockIdx.x;
  const float*  Xp  = dir ? XpB : XpF;
  const __bf16* Whh = dir ? WhhB : WhhF;
  const int dofs = dir ? 128 : 0;
  const int tid = threadIdx.x, lane = tid & 31, wave = tid >> 5;
  const int hi8 = (lane & 16) ? 8 : 0;
  const int col = lane & 15;

  for (int i = tid; i < 64 * 128; i += 512) { cst[i] = 0.f; hsh[i] = (__bf16)0.f; }
  __syncthreads();

  for (int t = 0; t < S; ++t) {
    const int s = dir ? (S - 1 - t) : t;
    v8f acc[2][4];
#pragma unroll
    for (int i = 0; i < 2; ++i) {
      const int p  = wave * 2 + i;          // 32 (m,u) tile pairs
      const int m0 = (p >> 3) * 16;
      const int u0 = (p & 7) * 16;
#pragma unroll
      for (int g = 0; g < 4; ++g) {
        const int n = g * 128 + u0 + col;
        const float* xp = Xp + ((size_t)(s * 64 + m0 + hi8)) * 512 + n;
#pragma unroll
        for (int v = 0; v < 8; ++v) acc[i][g][v] = xp[(size_t)v * 512];
      }
      for (int k0 = 0; k0 < 128; k0 += 32) {
        v16bf a = load_afrag(hsh, 128, m0, k0, lane);
#pragma unroll
        for (int g = 0; g < 4; ++g) {
          v16bf b = load_bfrag(Whh, 128, g * 128 + u0, k0, lane);
          acc[i][g] = WMMA_BF16(a, b, acc[i][g]);
        }
      }
    }
    __syncthreads();   // all reads of hsh complete
#pragma unroll
    for (int i = 0; i < 2; ++i) {
      const int p  = wave * 2 + i;
      const int m0 = (p >> 3) * 16;
      const int u0 = (p & 7) * 16;
      const int u  = u0 + col;
#pragma unroll
      for (int v = 0; v < 8; ++v) {
        const int b = m0 + hi8 + v;
        const float ig = sigf(acc[i][0][v]);
        const float fg = sigf(acc[i][1][v]);
        const float gg = tanhf(acc[i][2][v]);
        const float og = sigf(acc[i][3][v]);
        const float cn = fg * cst[b * 128 + u] + ig * gg;
        const float hn = og * tanhf(cn);
        cst[b * 128 + u] = cn;
        hsh[b * 128 + u] = (__bf16)hn;
        Xnext[((size_t)(s * 64 + b)) * 256 + dofs + u] = (__bf16)hn;
      }
    }
    __syncthreads();
  }
}

// enc repack: X3[s][b][d] -> encBF/encF rows b*127+s
__global__ void k_repack_enc(const __bf16* __restrict__ X3,
                             __bf16* __restrict__ encBF, float* __restrict__ encF) {
  int idx = blockIdx.x * blockDim.x + threadIdx.x;
  const int total = 8128 * 256;
  if (idx >= total) return;
  int d = idx % 256; int r = idx / 256;
  int b = r / 127, s = r % 127;
  __bf16 v = X3[((size_t)s * 64 + b) * 256 + d];
  encBF[idx] = v;
  encF[idx]  = (float)v;
}

// gather decoder input embeddings (teacher forcing) as bf16 rows t*64+b
__global__ void k_embtok(const int* __restrict__ y, const float* __restrict__ emb,
                         __bf16* __restrict__ etok) {
  int idx = blockIdx.x * blockDim.x + threadIdx.x;
  const int total = 100 * 64 * 64;
  if (idx >= total) return;
  int e = idx % 64; int r = idx / 64;
  int b = r % 64; int tt = r / 64;
  int tok = (tt == 0) ? 1 : y[b * 100 + tt - 1];
  etok[idx] = (__bf16)emb[(size_t)tok * 64 + e];
}

__global__ void k_init_out(float* __restrict__ out) {
  int idx = blockIdx.x * blockDim.x + threadIdx.x;
  if (idx >= 64 * 90) return;
  int b = idx / 90, v = idx % 90;
  out[((size_t)b * 101) * 90 + v] = 1.0f;   // float(BOS)
}

// ---------------------------------------------------------------------------
// Persistent attention decoder: 100 serial steps, one workgroup, 16 waves.
// ---------------------------------------------------------------------------
struct DecArgs {
  const float*  embProj;  // [100*64, 1024] = emb@Wih^T + b (precomputed)
  const __bf16* Whh;      // dec_Whh bf16 [1024,256]
  const __bf16* attWh;    // bf16 [128,256]
  const float*  attV;     // [128]
  const float*  encProj;  // [64*127, 128] (incl att_b)
  const float*  encF;     // [64*127, 256] f32
  const __bf16* combW;    // bf16 [128,512]
  const float*  combB;    // [128]
  const __bf16* outW;     // bf16 [96,128] zero-padded from [90,128]
  const float*  outB;     // [90]
  const float*  h0;       // [64,256]
  float* out;             // [64,101,90]
  float* hw;              // [64,128]
  float* alpha;           // [64,128] (slot 127 zeroed)
  float* cprev;           // [64,256]  (cell state <- context, per reference)
  __bf16* hA; __bf16* hB; // double-buffered h, bf16 [64,256]
  __bf16* concatb;        // [64,512]
  __bf16* attnb;          // [64,128]
};

__global__ __launch_bounds__(512) void k_decoder(DecArgs A) {
  const int T = 100;
  const int tid = threadIdx.x, lane = tid & 31, wave = tid >> 5;
  const int hi8 = (lane & 16) ? 8 : 0;
  const int col = lane & 15;

  for (int i = tid; i < 64 * 256; i += 512) { A.hA[i] = (__bf16)A.h0[i]; A.cprev[i] = 0.f; }
  __syncthreads();

  for (int t = 0; t < T; ++t) {
    const __bf16* hR = (t & 1) ? A.hB : A.hA;
    __bf16*       hW = (t & 1) ? A.hA : A.hB;

    // ---- Phase A: gates[64,1024] = embProj[t] + h @ dec_Whh^T; LSTM cell ----
    for (int i = 0; i < 4; ++i) {
      const int p  = wave * 4 + i;          // 64 (m,u) pairs
      const int m0 = (p >> 4) * 16;
      const int u0 = (p & 15) * 16;
      v8f acc[4];
#pragma unroll
      for (int g = 0; g < 4; ++g) {
        const float* xp = A.embProj + ((size_t)(t * 64 + m0 + hi8)) * 1024 + g * 256 + u0 + col;
#pragma unroll
        for (int v = 0; v < 8; ++v) acc[g][v] = xp[(size_t)v * 1024];
      }
      for (int k0 = 0; k0 < 256; k0 += 32) {
        v16bf a = load_afrag(hR, 256, m0, k0, lane);
#pragma unroll
        for (int g = 0; g < 4; ++g) {
          v16bf b = load_bfrag(A.Whh, 256, g * 256 + u0, k0, lane);
          acc[g] = WMMA_BF16(a, b, acc[g]);
        }
      }
      const int u = u0 + col;
#pragma unroll
      for (int v = 0; v < 8; ++v) {
        const int b = m0 + hi8 + v;
        const float cn = sigf(acc[1][v]) * A.cprev[b * 256 + u] +
                         sigf(acc[0][v]) * tanhf(acc[2][v]);
        const float hn = sigf(acc[3][v]) * tanhf(cn);
        hW[b * 256 + u] = (__bf16)hn;     // cell state for next step is the context (ref quirk)
      }
    }
    __syncthreads();

    // ---- Phase B: hw = h @ att_Wh^T  (M=64,N=128,K=256) ----
    for (int i = 0; i < 2; ++i) {
      const int idx = wave * 2 + i;        // 32 tiles
      const int m0 = (idx >> 3) * 16, n0 = (idx & 7) * 16;
      v8f acc = {};
      for (int k0 = 0; k0 < 256; k0 += 32)
        acc = WMMA_BF16(load_afrag(hW, 256, m0, k0, lane),
                        load_bfrag(A.attWh, 256, n0, k0, lane), acc);
      const int n = n0 + col;
#pragma unroll
      for (int v = 0; v < 8; ++v) A.hw[(m0 + hi8 + v) * 128 + n] = acc[v];
    }
    __syncthreads();

    // ---- Phase C: scores[b,s] = sum_a tanh(hw + encProj) * v ; softmax ----
    for (int pid = tid; pid < 64 * 127; pid += 512) {
      const int b = pid / 127, s = pid % 127;
      const float* ep  = A.encProj + ((size_t)b * 127 + s) * 128;
      const float* hwp = A.hw + b * 128;
      float sc = 0.f;
      for (int a2 = 0; a2 < 128; ++a2) sc += tanhf(hwp[a2] + ep[a2]) * A.attV[a2];
      A.alpha[b * 128 + s] = sc;
    }
    __syncthreads();
    if (tid < 64) {
      float* al = A.alpha + tid * 128;
      float mx = -3.0e38f;
      for (int s = 0; s < 127; ++s) mx = fmaxf(mx, al[s]);
      float sum = 0.f;
      for (int s = 0; s < 127; ++s) { float e = __expf(al[s] - mx); al[s] = e; sum += e; }
      const float inv = 1.f / sum;
      for (int s = 0; s < 127; ++s) al[s] *= inv;
      al[127] = 0.f;
    }
    __syncthreads();

    // ---- Phase D: context = alpha @ enc (batched); build concat(ctx, h) ----
    for (int idx = tid; idx < 64 * 256; idx += 512) {
      const int b = idx >> 8, d = idx & 255;
      const float* ef = A.encF + ((size_t)b * 127) * 256 + d;
      const float* al = A.alpha + b * 128;
      float cx = 0.f;
      for (int s2 = 0; s2 < 127; ++s2) cx += al[s2] * ef[(size_t)s2 * 256];
      A.cprev[idx] = cx;                       // feeds back as cell state
      A.concatb[b * 512 + d]       = (__bf16)cx;
      A.concatb[b * 512 + 256 + d] = hW[idx];
    }
    __syncthreads();

    // ---- Phase E: attn = relu(concat @ comb_W^T + comb_b)  (K=512) ----
    for (int i = 0; i < 2; ++i) {
      const int idx = wave * 2 + i;
      const int m0 = (idx >> 3) * 16, n0 = (idx & 7) * 16;
      v8f acc = {};
      for (int k0 = 0; k0 < 512; k0 += 32)
        acc = WMMA_BF16(load_afrag(A.concatb, 512, m0, k0, lane),
                        load_bfrag(A.combW, 512, n0, k0, lane), acc);
      const int n = n0 + col;
#pragma unroll
      for (int v = 0; v < 8; ++v) {
        const float r = acc[v] + A.combB[n];
        A.attnb[(m0 + hi8 + v) * 128 + n] = (__bf16)(r > 0.f ? r : 0.f);
      }
    }
    __syncthreads();

    // ---- Phase F: logits = attn @ out_W^T + out_b (N padded to 96) ----
    for (int idx = wave; idx < 24; idx += 16) {
      const int m0 = (idx / 6) * 16, n0 = (idx % 6) * 16;
      v8f acc = {};
      for (int k0 = 0; k0 < 128; k0 += 32)
        acc = WMMA_BF16(load_afrag(A.attnb, 128, m0, k0, lane),
                        load_bfrag(A.outW, 128, n0, k0, lane), acc);
      const int n = n0 + col;
      if (n < 90) {
        const float bv = A.outB[n];
#pragma unroll
        for (int v = 0; v < 8; ++v)
          A.out[((size_t)(m0 + hi8 + v) * 101 + t + 1) * 90 + n] = acc[v] + bv;
      }
    }
    __syncthreads();
  }
}

// ---------------------------------------------------------------------------
// Host orchestration
// ---------------------------------------------------------------------------
extern "C" void kernel_launch(void* const* d_in, const int* in_sizes, int n_in,
                              void* d_out, int out_size, void* d_ws, size_t ws_size,
                              hipStream_t stream) {
  const float* x       = (const float*)d_in[0];
  const int*   y       = (const int*)d_in[1];
  const float* h0      = (const float*)d_in[2];
  const float* conv1_w = (const float*)d_in[3];
  const float* conv1_b = (const float*)d_in[4];
  const float* conv2_w = (const float*)d_in[5];
  const float* conv2_b = (const float*)d_in[6];
  const float* conv3_w = (const float*)d_in[7];
  const float* conv3_b = (const float*)d_in[8];
  // enc params: base 9, layout l in {0,1,2}, dir in {f=0,b=1}: Wih, Whh, b
  auto encP = [&](int l, int d, int which) { return (const float*)d_in[9 + l * 6 + d * 3 + which]; };
  const float* emb    = (const float*)d_in[27];
  const float* decWih = (const float*)d_in[28];
  const float* decWhh = (const float*)d_in[29];
  const float* decB   = (const float*)d_in[30];
  const float* attWh  = (const float*)d_in[31];
  const float* attWs  = (const float*)d_in[32];
  const float* attV   = (const float*)d_in[33];
  const float* attB   = (const float*)d_in[34];
  const float* combW  = (const float*)d_in[35];
  const float* combB  = (const float*)d_in[36];
  const float* outW   = (const float*)d_in[37];
  const float* outB   = (const float*)d_in[38];
  float* out = (float*)d_out;

  // workspace allocator (256B aligned)
  char* base = (char*)d_ws;
  size_t off = 0;
  auto alloc = [&](size_t bytes) -> void* {
    void* p = base + off;
    off = (off + bytes + 255) & ~(size_t)255;
    return p;
  };

  float*  c1      = (float*)alloc((size_t)64 * 8 * 15 * 511 * 4);
  float*  c2      = (float*)alloc((size_t)64 * 32 * 10 * 508 * 4);
  float*  p1      = (float*)alloc((size_t)64 * 32 * 2 * 254 * 4);
  float*  c3      = (float*)alloc((size_t)64 * 64 * 2 * 254 * 4);
  __bf16* X0      = (__bf16*)alloc((size_t)127 * 64 * 128 * 2);
  __bf16* X1      = (__bf16*)alloc((size_t)127 * 64 * 256 * 2);
  __bf16* X2      = (__bf16*)alloc((size_t)127 * 64 * 256 * 2);
  __bf16* X3      = (__bf16*)alloc((size_t)127 * 64 * 256 * 2);
  float*  XpF     = (float*)alloc((size_t)8128 * 512 * 4);
  float*  XpB     = (float*)alloc((size_t)8128 * 512 * 4);
  __bf16* encBF   = (__bf16*)alloc((size_t)8128 * 256 * 2);
  float*  encF    = (float*)alloc((size_t)8128 * 256 * 4);
  float*  encProj = (float*)alloc((size_t)8128 * 128 * 4);
  __bf16* etok    = (__bf16*)alloc((size_t)6400 * 64 * 2);
  float*  embProj = (float*)alloc((size_t)6400 * 1024 * 4);
  float*  d_hw    = (float*)alloc((size_t)64 * 128 * 4);
  float*  d_alpha = (float*)alloc((size_t)64 * 128 * 4);
  float*  d_cprev = (float*)alloc((size_t)64 * 256 * 4);
  __bf16* d_hA    = (__bf16*)alloc((size_t)64 * 256 * 2);
  __bf16* d_hB    = (__bf16*)alloc((size_t)64 * 256 * 2);
  __bf16* d_cat   = (__bf16*)alloc((size_t)64 * 512 * 2);
  __bf16* d_attn  = (__bf16*)alloc((size_t)64 * 128 * 2);
  // bf16 weight mirrors (L2-resident; converted once per launch)
  __bf16* wWih[3][2]; __bf16* wWhh[3][2];
  for (int l = 0; l < 3; ++l)
    for (int d = 0; d < 2; ++d) {
      wWih[l][d] = (__bf16*)alloc((size_t)512 * (l == 0 ? 128 : 256) * 2);
      wWhh[l][d] = (__bf16*)alloc((size_t)512 * 128 * 2);
    }
  __bf16* wDecWhh = (__bf16*)alloc((size_t)1024 * 256 * 2);
  __bf16* wDecWih = (__bf16*)alloc((size_t)1024 * 64 * 2);
  __bf16* wAttWh  = (__bf16*)alloc((size_t)128 * 256 * 2);
  __bf16* wAttWs  = (__bf16*)alloc((size_t)128 * 256 * 2);
  __bf16* wCombW  = (__bf16*)alloc((size_t)128 * 512 * 2);
  __bf16* wOutW   = (__bf16*)alloc((size_t)96 * 128 * 2);   // padded rows 90..95 = 0
  (void)ws_size; (void)in_sizes; (void)n_in; (void)out_size;

  const int TPB = 256;
  auto blocks = [&](long long n) { return (unsigned)((n + TPB - 1) / TPB); };
  auto cvt = [&](const float* s, __bf16* d, int n) {
    k_cvt_bf16<<<blocks(n), TPB, 0, stream>>>(s, d, n);
  };

  // --- weight conversion pass (tiny; overlaps nothing, L2-warms weights) ---
  for (int l = 0; l < 3; ++l)
    for (int d = 0; d < 2; ++d) {
      cvt(encP(l, d, 0), wWih[l][d], 512 * (l == 0 ? 128 : 256));
      cvt(encP(l, d, 1), wWhh[l][d], 512 * 128);
    }
  cvt(decWhh, wDecWhh, 1024 * 256);
  cvt(decWih, wDecWih, 1024 * 64);
  cvt(attWh, wAttWh, 128 * 256);
  cvt(attWs, wAttWs, 128 * 256);
  cvt(combW, wCombW, 128 * 512);
  k_cvt_bf16_pad<<<blocks(96 * 128), TPB, 0, stream>>>(outW, wOutW, 90, 128, 96);

  k_init_out<<<blocks(64 * 90), TPB, 0, stream>>>(out);
  k_conv1<<<blocks((long long)64 * 8 * 15 * 511), TPB, 0, stream>>>(x, conv1_w, conv1_b, c1);
  k_conv2<<<blocks((long long)64 * 32 * 10 * 508), TPB, 0, stream>>>(c1, conv2_w, conv2_b, c2);
  k_pool1<<<blocks((long long)64 * 32 * 2 * 254), TPB, 0, stream>>>(c2, p1);
  k_conv3<<<blocks((long long)64 * 64 * 2 * 254), TPB, 0, stream>>>(p1, conv3_w, conv3_b, c3);
  k_pool2_repack<<<blocks((long long)64 * 64 * 2 * 127), TPB, 0, stream>>>(c3, X0);

  // 3 bidirectional LSTM layers
  const __bf16* Xin[3]  = { X0, X1, X2 };
  __bf16*       Xout[3] = { X1, X2, X3 };
  for (int l = 0; l < 3; ++l) {
    const int din = (l == 0) ? 128 : 256;
    const unsigned gtiles = (8128 / 16) * (512 / 16);
    k_gemm_bf16<<<gtiles, 32, 0, stream>>>(Xin[l], wWih[l][0], encP(l, 0, 2), XpF, 512, din);
    k_gemm_bf16<<<gtiles, 32, 0, stream>>>(Xin[l], wWih[l][1], encP(l, 1, 2), XpB, 512, din);
    k_lstm_layer<<<2, 512, 0, stream>>>(XpF, XpB, wWhh[l][0], wWhh[l][1], Xout[l]);
  }

  k_repack_enc<<<blocks((long long)8128 * 256), TPB, 0, stream>>>(X3, encBF, encF);
  k_gemm_bf16<<<(8128 / 16) * (128 / 16), 32, 0, stream>>>(encBF, wAttWs, attB, encProj, 128, 256);
  k_embtok<<<blocks((long long)100 * 64 * 64), TPB, 0, stream>>>(y, emb, etok);
  k_gemm_bf16<<<(6400 / 16) * (1024 / 16), 32, 0, stream>>>(etok, wDecWih, decB, embProj, 1024, 64);

  DecArgs DA;
  DA.embProj = embProj; DA.Whh = wDecWhh; DA.attWh = wAttWh; DA.attV = attV;
  DA.encProj = encProj; DA.encF = encF; DA.combW = wCombW; DA.combB = combB;
  DA.outW = wOutW; DA.outB = outB; DA.h0 = h0; DA.out = out;
  DA.hw = d_hw; DA.alpha = d_alpha; DA.cprev = d_cprev;
  DA.hA = d_hA; DA.hB = d_hB; DA.concatb = d_cat; DA.attnb = d_attn;
  k_decoder<<<1, 512, 0, stream>>>(DA);
}
